// TransformerRecommenderWithMixedAttention_65077344469582
// MI455X (gfx1250) — matrix operand order repaired and checked by reading, not compiled, not run on hardware
//
#include <hip/hip_runtime.h>

// ---------------------------------------------------------------------------
// Types / WMMA helpers (CDNA5 / gfx1250, wave32)
// ---------------------------------------------------------------------------
typedef __attribute__((ext_vector_type(16))) _Float16 v16h;
typedef __attribute__((ext_vector_type(8)))  _Float16 v8h;
typedef __attribute__((ext_vector_type(8)))  float    v8f;

#define DEVINL __device__ __forceinline__

// Build a 16xhalf fragment from two contiguous 16B chunks.
// A-fragment (16x32, MxK) and B-fragment (32x16, KxN): this code loads 16
// contiguous K values per lane from (lane>>4)*16 (same order in A and B) -> p1 = p0 + 8.
DEVINL v16h frag16(const _Float16* p0, const _Float16* p1) {
  v8h a = *(const v8h*)p0;
  v8h b = *(const v8h*)p1;
  return __builtin_shufflevector(a, b, 0,1,2,3,4,5,6,7,8,9,10,11,12,13,14,15);
}

DEVINL v8f wmma_f16(v16h a, v16h b, v8f c) {
  // D = A(16x32 f16) * B(32x16 f16) + C(16x16 f32)
  return __builtin_amdgcn_wmma_f32_16x16x32_f16(false, a, false, b,
                                                (short)0, c, false, false);
}

// ---------------------------------------------------------------------------
// WMMA GEMM:  C[M,N] = A[M,K] (f16, row-major) * W[N,K]^T (f16, row-major)
//             + bias[N], optional ReLU; writes f32 and/or f16 outputs.
// Wave tile 32x64 (2 A frags x 4 B frags -> 8 WMMAs / 32-k step);
// block = 8 waves (4 in M, 2 in N) -> 128x128 macro tile.
// Requires M%128==0, N%128==0, K%32==0 (true for all GEMMs here).
// ---------------------------------------------------------------------------
__global__ void __launch_bounds__(256)
k_gemm(const _Float16* __restrict__ A, const _Float16* __restrict__ Wt,
       const float* __restrict__ bias, float* __restrict__ o32,
       _Float16* __restrict__ o16, int M, int N, int K, int ldc, int relu)
{
  const int lane = threadIdx.x & 31;
  const int wave = threadIdx.x >> 5;
  const int lh = lane & 15, hb = lane >> 4;
  const int m0 = blockIdx.y * 128 + (wave & 3) * 32;
  const int n0 = blockIdx.x * 128 + (wave >> 2) * 64;
  if (m0 >= M) return;   // wave-uniform

  const _Float16* pa0 = A  + (size_t)(m0 + lh) * K + hb * 16;
  const _Float16* pa1 = pa0 + (size_t)16 * K;
  const _Float16* pb0 = Wt + (size_t)(n0 + lh) * K + hb * 16;
  const _Float16* pb1 = pb0 + (size_t)16 * K;
  const _Float16* pb2 = pb1 + (size_t)16 * K;
  const _Float16* pb3 = pb2 + (size_t)16 * K;

  v8f z = {};
  v8f acc[2][4];
  for (int i = 0; i < 2; ++i)
    for (int j = 0; j < 4; ++j) acc[i][j] = z;

  for (int k0 = 0; k0 < K; k0 += 32) {
    // locality 3 -> WGP-scope prefetch (pull into near caches)
    __builtin_prefetch(pa0 + k0 + 64, 0, 3);
    v16h a0 = frag16(pa0 + k0, pa0 + k0 + 8);
    v16h a1 = frag16(pa1 + k0, pa1 + k0 + 8);
    v16h b0 = frag16(pb0 + k0, pb0 + k0 + 8);
    v16h b1 = frag16(pb1 + k0, pb1 + k0 + 8);
    v16h b2 = frag16(pb2 + k0, pb2 + k0 + 8);
    v16h b3 = frag16(pb3 + k0, pb3 + k0 + 8);
    acc[0][0] = wmma_f16(a0, b0, acc[0][0]);
    acc[0][1] = wmma_f16(a0, b1, acc[0][1]);
    acc[0][2] = wmma_f16(a0, b2, acc[0][2]);
    acc[0][3] = wmma_f16(a0, b3, acc[0][3]);
    acc[1][0] = wmma_f16(a1, b0, acc[1][0]);
    acc[1][1] = wmma_f16(a1, b1, acc[1][1]);
    acc[1][2] = wmma_f16(a1, b2, acc[1][2]);
    acc[1][3] = wmma_f16(a1, b3, acc[1][3]);
  }
  for (int i = 0; i < 2; ++i)
    for (int j = 0; j < 4; ++j) {
      const int col = n0 + j * 16 + lh;
      const float bv = bias ? bias[col] : 0.0f;
      const int rowb = m0 + i * 16 + hb * 8;
      for (int r = 0; r < 8; ++r) {
        float v = acc[i][j][r] + bv;
        if (relu) v = fmaxf(v, 0.0f);
        const size_t off = (size_t)(rowb + r) * ldc + col;
        if (o32) o32[off] = v;
        if (o16) o16[off] = (_Float16)v;
      }
    }
}

// ---------------------------------------------------------------------------
// Fused mixed-mask attention (flash style, online softmax).
// qkv f16 [S,B,1536]: Q at col 0, K at 512, V at 1024 (head h -> +h*64).
// vt  f16 [B,H,64,S]: pre-transposed V so PV B-fragments are contiguous.
// One wave handles (b, h, 16 q rows); loops 32 keys at a time:
//   4 WMMAs QK^T + mask/softmax + LDS P-transpose + 4 WMMAs PV.
// ---------------------------------------------------------------------------
__global__ void __launch_bounds__(128)
k_attn(const _Float16* __restrict__ qkv, const _Float16* __restrict__ vt,
       _Float16* __restrict__ out, int S, int B)
{
  __shared__ _Float16 ldsP[4][16 * 32];   // per-wave P staging (f16)
  const int lane = threadIdx.x & 31;
  const int wave = threadIdx.x >> 5;
  const int lh = lane & 15, hb = lane >> 4;
  const int bh = blockIdx.y;
  const int b = bh >> 3, h = bh & 7;      // H == 8
  const int m0 = (blockIdx.x * 4 + wave) * 16;

  const size_t tokStride = (size_t)B * 1536;
  const _Float16* Qbase = qkv + (size_t)b * 1536 + h * 64;
  const _Float16* Kbase = Qbase + 512;
  const _Float16* Vt = vt + (size_t)(b * 8 + h) * 64 * S;
  _Float16* myP = &ldsP[wave][0];

  v16h aq0, aq1;   // Q tile 16x64 as two A fragments
  {
    const _Float16* q = Qbase + (size_t)(m0 + lh) * tokStride + hb * 16;
    aq0 = frag16(q, q + 8);
    aq1 = frag16(q + 32, q + 40);
  }

  float mrun[8], lrun[8];
  v8f z = {};
  v8f o0 = z, o1 = z, o2 = z, o3 = z;
  for (int r = 0; r < 8; ++r) { mrun[r] = -1e30f; lrun[r] = 0.0f; }

  for (int kb = 0; kb < S; kb += 32) {
    // ---- scores: two 16x16 tiles over hd=64 (4 WMMAs) ----
    v8f s0 = z, s1 = z;
    {
      const _Float16* kr0 = Kbase + (size_t)(kb + lh) * tokStride + hb * 16;
      s0 = wmma_f16(aq0, frag16(kr0, kr0 + 8), s0);
      s0 = wmma_f16(aq1, frag16(kr0 + 32, kr0 + 40), s0);
      const _Float16* kr1 = kr0 + (size_t)16 * tokStride;
      s1 = wmma_f16(aq0, frag16(kr1, kr1 + 8), s1);
      s1 = wmma_f16(aq1, frag16(kr1 + 32, kr1 + 40), s1);
    }
    // ---- mask + scale; every 32-key block contains a j%10==0 column so
    //      every row's max stays finite (no inf/nan paths) ----
    float p0[8], p1[8], lmax[8];
    const int c0 = kb + lh, c1 = kb + 16 + lh;
    const bool kc0 = (c0 % 10) == 0, kc1 = (c1 % 10) == 0;
    for (int r = 0; r < 8; ++r) {
      const int row = m0 + r + hb * 8;
      int d0 = row - c0; if (d0 < 0) d0 = -d0;
      int d1 = row - c1; if (d1 < 0) d1 = -d1;
      const float v0 = (d0 <= 5 || kc0) ? s0[r] * 0.125f : -1e30f;
      const float v1 = (d1 <= 5 || kc1) ? s1[r] * 0.125f : -1e30f;
      p0[r] = v0; p1[r] = v1;
      lmax[r] = fmaxf(v0, v1);
    }
    for (int off = 1; off < 16; off <<= 1)
      for (int r = 0; r < 8; ++r)
        lmax[r] = fmaxf(lmax[r], __shfl_xor(lmax[r], off));
    float lsum[8];
    for (int r = 0; r < 8; ++r) {
      const float mn = fmaxf(mrun[r], lmax[r]);
      const float cf = __expf(mrun[r] - mn);
      mrun[r] = mn;
      const float e0 = __expf(p0[r] - mn);
      const float e1 = __expf(p1[r] - mn);
      p0[r] = e0; p1[r] = e1;
      lsum[r] = e0 + e1;
      lrun[r] *= cf;
      o0[r] *= cf; o1[r] *= cf; o2[r] *= cf; o3[r] *= cf;
    }
    for (int off = 1; off < 16; off <<= 1)
      for (int r = 0; r < 8; ++r)
        lsum[r] += __shfl_xor(lsum[r], off);
    for (int r = 0; r < 8; ++r) lrun[r] += lsum[r];

    // ---- transpose P (C-layout -> A-layout) through wave-private LDS ----
    for (int r = 0; r < 8; ++r) {
      const int rr = r + hb * 8;
      myP[rr * 32 + lh]      = (_Float16)p0[r];
      myP[rr * 32 + 16 + lh] = (_Float16)p1[r];
    }
    asm volatile("s_wait_dscnt 0" ::: "memory");
    v16h ap;
    {
      const _Float16* pr = myP + lh * 32 + hb * 16;
      ap = frag16(pr, pr + 8);
    }
    // ---- PV: P(16x32) x V(32x64) -> 4 WMMAs, contiguous Vt rows ----
    {
      const _Float16* v0p = Vt + (size_t)lh * S + kb + hb * 16;
      o0 = wmma_f16(ap, frag16(v0p, v0p + 8), o0);
      const _Float16* v1p = v0p + (size_t)16 * S;
      o1 = wmma_f16(ap, frag16(v1p, v1p + 8), o1);
      const _Float16* v2p = v1p + (size_t)16 * S;
      o2 = wmma_f16(ap, frag16(v2p, v2p + 8), o2);
      const _Float16* v3p = v2p + (size_t)16 * S;
      o3 = wmma_f16(ap, frag16(v3p, v3p + 8), o3);
    }
  }
  for (int r = 0; r < 8; ++r) {
    const int row = m0 + r + hb * 8;
    const float inv = 1.0f / lrun[r];
    _Float16* orow = out + ((size_t)row * B + b) * 512 + h * 64;
    orow[lh]      = (_Float16)(o0[r] * inv);
    orow[16 + lh] = (_Float16)(o1[r] * inv);
    orow[32 + lh] = (_Float16)(o2[r] * inv);
    orow[48 + lh] = (_Float16)(o3[r] * inv);
  }
}

// ---------------------------------------------------------------------------
// Small supporting kernels
// ---------------------------------------------------------------------------
__global__ void __launch_bounds__(256)
k_cvt(const float* __restrict__ in, _Float16* __restrict__ out, int n) {
  const int i = blockIdx.x * 256 + threadIdx.x;
  if (i < n) out[i] = (_Float16)in[i];
}

// vt[b][h][d][s] = qkv[s][b][1024 + h*64 + d]
__global__ void __launch_bounds__(256)
k_vt(const _Float16* __restrict__ qkv, _Float16* __restrict__ vt, int S, int B) {
  const int idx = blockIdx.x * 256 + threadIdx.x;
  const int s = idx % S;
  const int d = (idx / S) & 63;
  const int h = (idx / (S * 64)) & 7;
  const int b = idx / (S * 64 * 8);
  vt[idx] = qkv[((size_t)s * B + b) * 1536 + 1024 + h * 64 + d];
}

// NOTE: reference does x + pos[:x.shape[1]] with x of shape (S,B,E), so pos
// is indexed by the *batch* slot and broadcast over S.  Mirrored here.
__global__ void __launch_bounds__(256)
k_addpos(const float* __restrict__ X, const float* __restrict__ pos,
         float* __restrict__ o32, _Float16* __restrict__ o16, int B) {
  const int idx = blockIdx.x * 256 + threadIdx.x;
  const int e = idx & 511;
  const int m = idx >> 9;
  const int b = m % B;
  const float v = X[idx] + pos[b * 512 + e];
  o32[idx] = v;
  o16[idx] = (_Float16)v;
}

// out = LayerNorm(X + R) * g + be ; one wave per 512-wide row
__global__ void __launch_bounds__(256)
k_ln(const float* __restrict__ X, const float* __restrict__ R,
     const float* __restrict__ g, const float* __restrict__ be,
     float* __restrict__ o32, _Float16* __restrict__ o16, int M) {
  const int wave = threadIdx.x >> 5, lane = threadIdx.x & 31;
  const int row = blockIdx.x * 8 + wave;
  if (row >= M) return;
  const float* x = X + (size_t)row * 512;
  const float* rr = R + (size_t)row * 512;
  float v[16], s = 0.0f;
  for (int t = 0; t < 16; ++t) { v[t] = x[lane + 32 * t] + rr[lane + 32 * t]; s += v[t]; }
  for (int off = 1; off < 32; off <<= 1) s += __shfl_xor(s, off);
  const float mu = s * (1.0f / 512.0f);
  float vs = 0.0f;
  for (int t = 0; t < 16; ++t) { const float d = v[t] - mu; vs += d * d; }
  for (int off = 1; off < 32; off <<= 1) vs += __shfl_xor(vs, off);
  const float rstd = rsqrtf(vs * (1.0f / 512.0f) + 1e-5f);
  for (int t = 0; t < 16; ++t) {
    const int e = lane + 32 * t;
    const float y = (v[t] - mu) * rstd * g[e] + be[e];
    o32[(size_t)row * 512 + e] = y;
    o16[(size_t)row * 512 + e] = (_Float16)y;
  }
}

// im2col for the stride-2 conv1d: A[m, i*3+t] = emb[ids[b, 2s-1+t], i]
// (k = i*3+t makes conv_w (E,E,3) the GEMM weight with no rearrangement)
__global__ void __launch_bounds__(256)
k_gather(const int* __restrict__ ids, const float* __restrict__ emb,
         _Float16* __restrict__ A, int B, int L) {
  const int m = blockIdx.x;
  const int s = m / B, b = m % B;
  _Float16* arow = A + (size_t)m * 1536;
  for (int t = 0; t < 3; ++t) {
    const int l = 2 * s - 1 + t;
    const bool valid = (l >= 0 && l < L);
    const int id = valid ? ids[b * L + l] : 0;
    const float* erow = emb + (size_t)id * 512;
    for (int i = threadIdx.x; i < 512; i += 256)
      arow[i * 3 + t] = valid ? (_Float16)erow[i] : (_Float16)0.0f;
  }
}

__global__ void __launch_bounds__(256)
k_concat(const float* __restrict__ a, const float* __restrict__ b2,
         _Float16* __restrict__ o) {
  const int idx = blockIdx.x * 256 + threadIdx.x;
  const int m = idx >> 10, c = idx & 1023;
  const float v = (c < 512) ? a[(size_t)m * 512 + c] : b2[(size_t)m * 512 + c - 512];
  o[idx] = (_Float16)v;
}

__global__ void __launch_bounds__(256)
k_maxpool(const float* __restrict__ d0, const float* __restrict__ d1,
          float* __restrict__ o, int S, int B) {
  const int idx = blockIdx.x * 256 + threadIdx.x;   // B*1024
  const int b = idx >> 10, f = idx & 1023;
  const float* src = (f < 512) ? d0 : d1;
  const int ff = f & 511;
  float m = -3.4e38f;
  for (int s = 0; s < S; ++s) m = fmaxf(m, src[((size_t)s * B + b) * 512 + ff]);
  o[idx] = m;
}

// FC head: hfc = relu(pooled @ fc1^T + b1) [16,20]; out = hfc @ fc2^T + b2 [16,2]
__global__ void __launch_bounds__(320)
k_fc(const float* __restrict__ pooled, const float* __restrict__ w1,
     const float* __restrict__ b1, const float* __restrict__ w2,
     const float* __restrict__ b2, float* __restrict__ out) {
  __shared__ float hf[16 * 20];
  const int tid = threadIdx.x;
  if (tid < 320) {
    const int bb = tid / 20, jj = tid % 20;
    float s = b1[jj];
    for (int k = 0; k < 1024; ++k) s += pooled[bb * 1024 + k] * w1[jj * 1024 + k];
    hf[tid] = fmaxf(s, 0.0f);
  }
  __syncthreads();
  if (tid < 32) {
    const int bb = tid >> 1, oo = tid & 1;
    float s = b2[oo];
    for (int k = 0; k < 20; ++k) s += hf[bb * 20 + k] * w2[oo * 20 + k];
    out[bb * 2 + oo] = s;
  }
}

// ---------------------------------------------------------------------------
// Host orchestration
// ---------------------------------------------------------------------------
extern "C" void kernel_launch(void* const* d_in, const int* in_sizes, int n_in,
                              void* d_out, int out_size, void* d_ws, size_t ws_size,
                              hipStream_t stream) {
  (void)in_sizes; (void)n_in; (void)out_size; (void)ws_size;
  constexpr int E = 512, H = 8, FFd = 2048, Bb = 16, L = 1024, S = 512;
  constexpr int M = S * Bb, K3 = 3 * E;

  auto F = [&](int i) { return (const float*)d_in[i]; };
  auto Iptr = [&](int i) { return (const int*)d_in[i]; };

  // jax pytree leaf order (dict keys sorted alphabetically):
  // params: adj_b, adj_w, conv_b, conv_w, dec[2][2]{18 leaves}, emb,
  //         enc[2][2]{13 leaves}, fc1_b, fc1_w, fc2_b, fc2_w; then target_seq, x
  const int IDX_ADJ_B = 0, IDX_ADJ_W = 1, IDX_CONV_B = 2, IDX_CONV_W = 3;
  const int IDX_DEC0 = 4, IDX_EMB = 76, IDX_ENC0 = 77;
  const int IDX_FC1_B = 129, IDX_FC1_W = 130, IDX_FC2_B = 131, IDX_FC2_W = 132;
  const int IDX_TGT = 133, IDX_X = 134;
  enum { D_CIN_B = 0, D_CIN_W, D_COUT_B, D_COUT_W, D_FF1B, D_FF1W, D_FF2B, D_FF2W,
         D_LN1B, D_LN1W, D_LN2B, D_LN2W, D_LN3B, D_LN3W,
         D_SIN_B, D_SIN_W, D_SOUT_B, D_SOUT_W };
  enum { E_IN_B = 0, E_IN_W, E_OUT_B, E_OUT_W, E_FF1B, E_FF1W, E_FF2B, E_FF2W,
         E_LN1B, E_LN1W, E_LN2B, E_LN2W, E_POS };

  char* wsp = (char*)d_ws;
  auto alloc = [&](size_t bytes) -> char* {
    char* r = wsp; wsp += (bytes + 255) & ~(size_t)255; return r;
  };

  _Float16* wf16  = (_Float16*)alloc((size_t)FFd * E * 2);     // weight cvt scratch
  _Float16* bigA  = (_Float16*)alloc((size_t)M * K3 * 2);      // im2col / concat
  float*    XE32  = (float*)alloc((size_t)M * E * 4);
  _Float16* XE16  = (_Float16*)alloc((size_t)M * E * 2);
  float*    TE32  = (float*)alloc((size_t)M * E * 4);
  _Float16* TE16  = (_Float16*)alloc((size_t)M * E * 2);
  float*    XA32  = (float*)alloc((size_t)M * E * 4);
  _Float16* XA16  = (_Float16*)alloc((size_t)M * E * 2);
  float*    XB32  = (float*)alloc((size_t)M * E * 4);
  _Float16* XB16  = (_Float16*)alloc((size_t)M * E * 2);
  float*    XP32  = (float*)alloc((size_t)M * E * 4);
  _Float16* XP16  = (_Float16*)alloc((size_t)M * E * 2);
  _Float16* qkv   = (_Float16*)alloc((size_t)M * K3 * 2);
  _Float16* vt    = (_Float16*)alloc((size_t)Bb * H * 64 * S * 2);
  _Float16* aout  = (_Float16*)alloc((size_t)M * E * 2);
  float*    tmp32 = (float*)alloc((size_t)M * E * 4);
  _Float16* ff1   = (_Float16*)alloc((size_t)M * FFd * 2);
  float*    enc0  = (float*)alloc((size_t)M * E * 4);
  float*    enc1  = (float*)alloc((size_t)M * E * 4);
  _Float16* ecomb = (_Float16*)alloc((size_t)M * E * 2);
  float*    dc0   = (float*)alloc((size_t)M * E * 4);
  float*    dc1   = (float*)alloc((size_t)M * E * 4);
  float*    pooled = (float*)alloc((size_t)Bb * 2 * E * 4);

  auto cvt = [&](const float* src, _Float16* dst, int n) {
    k_cvt<<<dim3((n + 255) / 256), dim3(256), 0, stream>>>(src, dst, n);
  };
  auto gemm = [&](const _Float16* A, const _Float16* Wt, const float* bias,
                  float* o32, _Float16* o16, int N, int K, int ldc, int relu) {
    k_gemm<<<dim3(N / 128, M / 128), dim3(256), 0, stream>>>(A, Wt, bias, o32, o16,
                                                             M, N, K, ldc, relu);
  };
  auto attn = [&]() {
    k_vt<<<dim3((Bb * H * 64 * S) / 256), dim3(256), 0, stream>>>(qkv, vt, S, Bb);
    k_attn<<<dim3(S / 64, Bb * H), dim3(128), 0, stream>>>(qkv, vt, aout, S, Bb);
  };
  auto lnorm = [&](const float* X, const float* R, const float* g, const float* be,
                   float* o32, _Float16* o16) {
    k_ln<<<dim3(M / 8), dim3(256), 0, stream>>>(X, R, g, be, o32, o16, M);
  };
  auto embed = [&](const int* ids, float* out32, _Float16* out16) {
    k_gather<<<dim3(M), dim3(256), 0, stream>>>(ids, F(IDX_EMB), bigA, Bb, L);
    cvt(F(IDX_CONV_W), wf16, E * K3);
    gemm(bigA, wf16, F(IDX_CONV_B), out32, out16, E, K3, E, 0);
  };
  auto flip32 = [&](float* c) { return (c == XA32) ? XB32 : XA32; };
  auto flip16 = [&](float* c) { return (c == XA32) ? XB16 : XA16; };

  auto encLayer = [&](int base, float*& c32, _Float16*& c16) {
    k_addpos<<<dim3(M * E / 256), dim3(256), 0, stream>>>(c32, F(base + E_POS),
                                                          XP32, XP16, Bb);
    cvt(F(base + E_IN_W), wf16, K3 * E);
    gemm(XP16, wf16, F(base + E_IN_B), nullptr, qkv, K3, E, K3, 0);
    attn();
    cvt(F(base + E_OUT_W), wf16, E * E);
    gemm(aout, wf16, F(base + E_OUT_B), tmp32, nullptr, E, E, E, 0);
    float* n32 = flip32(c32); _Float16* n16 = flip16(c32);
    lnorm(XP32, tmp32, F(base + E_LN1W), F(base + E_LN1B), n32, n16);
    c32 = n32; c16 = n16;
    cvt(F(base + E_FF1W), wf16, FFd * E);
    gemm(c16, wf16, F(base + E_FF1B), nullptr, ff1, FFd, E, FFd, 1);
    cvt(F(base + E_FF2W), wf16, E * FFd);
    gemm(ff1, wf16, F(base + E_FF2B), tmp32, nullptr, E, FFd, E, 0);
    n32 = flip32(c32); n16 = flip16(c32);
    lnorm(c32, tmp32, F(base + E_LN2W), F(base + E_LN2B), n32, n16);
    c32 = n32; c16 = n16;
  };

  auto decLayer = [&](int base, float*& c32, _Float16*& c16) {
    // self attention
    cvt(F(base + D_SIN_W), wf16, K3 * E);
    gemm(c16, wf16, F(base + D_SIN_B), nullptr, qkv, K3, E, K3, 0);
    attn();
    cvt(F(base + D_SOUT_W), wf16, E * E);
    gemm(aout, wf16, F(base + D_SOUT_B), tmp32, nullptr, E, E, E, 0);
    float* n32 = flip32(c32); _Float16* n16 = flip16(c32);
    lnorm(c32, tmp32, F(base + D_LN1W), F(base + D_LN1B), n32, n16);
    c32 = n32; c16 = n16;
    // cross attention: Q from x, K/V from enc_comb (same qkv buffer layout)
    cvt(F(base + D_CIN_W), wf16, K3 * E);
    gemm(c16, wf16, F(base + D_CIN_B), nullptr, qkv, E, E, K3, 0);
    gemm(ecomb, wf16 + (size_t)E * E, F(base + D_CIN_B) + E, nullptr,
         qkv + E, 2 * E, E, K3, 0);
    attn();
    cvt(F(base + D_COUT_W), wf16, E * E);
    gemm(aout, wf16, F(base + D_COUT_B), tmp32, nullptr, E, E, E, 0);
    n32 = flip32(c32); n16 = flip16(c32);
    lnorm(c32, tmp32, F(base + D_LN2W), F(base + D_LN2B), n32, n16);
    c32 = n32; c16 = n16;
    // feed-forward
    cvt(F(base + D_FF1W), wf16, FFd * E);
    gemm(c16, wf16, F(base + D_FF1B), nullptr, ff1, FFd, E, FFd, 1);
    cvt(F(base + D_FF2W), wf16, E * FFd);
    gemm(ff1, wf16, F(base + D_FF2B), tmp32, nullptr, E, FFd, E, 0);
    n32 = flip32(c32); n16 = flip16(c32);
    lnorm(c32, tmp32, F(base + D_LN3W), F(base + D_LN3B), n32, n16);
    c32 = n32; c16 = n16;
  };

  // ---- encoder ----
  embed(Iptr(IDX_X), XE32, XE16);
  float* encOut[2] = {enc0, enc1};
  for (int st = 0; st < 2; ++st) {
    float* c32 = XE32; _Float16* c16 = XE16;
    for (int ly = 0; ly < 2; ++ly)
      encLayer(IDX_ENC0 + (st * 2 + ly) * 13, c32, c16);
    (void)hipMemcpyAsync(encOut[st], c32, (size_t)M * E * 4,
                         hipMemcpyDeviceToDevice, stream);
  }
  // enc_comb = concat(enc_outs) @ adj^T + adj_b  (f16, K/V source for cattn)
  k_concat<<<dim3(M * 2 * E / 256), dim3(256), 0, stream>>>(enc0, enc1, bigA);
  cvt(F(IDX_ADJ_W), wf16, E * 2 * E);
  gemm(bigA, wf16, F(IDX_ADJ_B), nullptr, ecomb, E, 2 * E, E, 0);

  // ---- decoder ----
  embed(Iptr(IDX_TGT), TE32, TE16);
  float* decOut[2] = {dc0, dc1};
  for (int st = 0; st < 2; ++st) {
    float* c32 = TE32; _Float16* c16 = TE16;
    for (int ly = 0; ly < 2; ++ly)
      decLayer(IDX_DEC0 + (st * 2 + ly) * 18, c32, c16);
    (void)hipMemcpyAsync(decOut[st], c32, (size_t)M * E * 4,
                         hipMemcpyDeviceToDevice, stream);
  }

  // ---- head ----
  k_maxpool<<<dim3(Bb * 2 * E / 256), dim3(256), 0, stream>>>(dc0, dc1, pooled, S, Bb);
  k_fc<<<dim3(1), dim3(320), 0, stream>>>(pooled, F(IDX_FC1_W), F(IDX_FC1_B),
                                          F(IDX_FC2_W), F(IDX_FC2_B), (float*)d_out);
}
